// Rnn_56178172231863
// MI455X (gfx1250) — compile-verified, matrix operand order checked by reading, not run on hardware
//
#include <hip/hip_runtime.h>
#include <hip/hip_bf16.h>
#include <stdint.h>

// ---------------------------------------------------------------------------
// LSTM (B=64, T=256, H=1024) with done-resets, tanh output activation.
// bf16 WMMA (v_wmma_f32_16x16x32_bf16) for both the big input projection and
// the 256 sequential recurrent GEMM steps. The recurrent step stages the
// shared h-tile into LDS via the Tensor Data Mover (tensor_load_to_lds).
// ---------------------------------------------------------------------------

#define Bq 64
#define Tq 256
#define Hq 1024
#define G4 4096   // 4*H

// LDS h-stage: 16 rows of 1024 bf16, TDM pads 2 dwords every 256 dwords
// -> row stride 2064 bytes (1032 shorts), 8-byte gap after element 512.
#define HSTRIDE_B 2064
#define HSTAGE_SHORTS (16 * 1032)

typedef __attribute__((ext_vector_type(16))) __bf16 v16bf;
typedef __attribute__((ext_vector_type(8)))  float  v8f;
typedef __attribute__((ext_vector_type(4)))  unsigned int u32x4;
typedef __attribute__((ext_vector_type(8)))  int i32x8;
typedef __attribute__((ext_vector_type(4)))  int i32x4;

union FragU {
    v16bf v;
    struct { uint4 lo; uint4 hi; } q;
};

// f32 -> bf16 (round to nearest even), stored as raw u16
__device__ __forceinline__ unsigned short f2bf(float f) {
    uint32_t u = __float_as_uint(f);
    u += 0x7FFFu + ((u >> 16) & 1u);
    return (unsigned short)(u >> 16);
}

__device__ __forceinline__ float sigm(float x) {
    return 1.0f / (1.0f + __expf(-x));
}

// Load a 16-bit WMMA operand fragment (A or B) from global memory.
// Per the CDNA5 16-bit layout: lanes 0-15 hold K = k0+0..7 / k0+16..23,
// lanes 16-31 hold K = k0+8..15 / k0+24..31 (8 bf16 per 16B chunk).
__device__ __forceinline__ v16bf load_frag16(const unsigned short* mat,
                                             int row, int kPitch,
                                             int k0, int lane) {
    int ko = k0 + ((lane >> 4) << 3);
    const unsigned short* p = mat + (size_t)row * kPitch + ko;
    FragU f;
    f.q.lo = *reinterpret_cast<const uint4*>(p);        // K = ko .. ko+7
    f.q.hi = *reinterpret_cast<const uint4*>(p + 16);   // K = ko+16 .. ko+23
    return f.v;
}

// ---------------------------------------------------------------------------
// Phase 0a: elementwise f32 -> bf16 conversion
// ---------------------------------------------------------------------------
__global__ void cvt_bf16_kernel(const float* __restrict__ src,
                                unsigned short* __restrict__ dst, int n) {
    int i = blockIdx.x * blockDim.x + threadIdx.x;
    int stride = gridDim.x * blockDim.x;
    for (; i < n; i += stride) dst[i] = f2bf(src[i]);
}

// Phase 0b: init recurrent state: h0 -> bf16 ping buffer 0, c0 -> f32
__global__ void init_state_kernel(const float* __restrict__ h0,
                                  const float* __restrict__ c0,
                                  unsigned short* __restrict__ hbuf0,
                                  float* __restrict__ cbuf) {
    int i = blockIdx.x * blockDim.x + threadIdx.x;
    if (i < Bq * Hq) {
        hbuf0[i] = f2bf(h0[i]);
        cbuf[i]  = c0[i];
    }
}

// ---------------------------------------------------------------------------
// Phase 1: x_proj[T*B, 4H] = x_bf16 @ W_ih^T + (b_ih + b_hh), f32 out.
// One wave per 16x16 output tile, K=1024 -> 32 chained WMMAs.
// ---------------------------------------------------------------------------
__global__ __launch_bounds__(128)
void xproj_gemm_kernel(const unsigned short* __restrict__ xb,
                       const unsigned short* __restrict__ wih,
                       const float* __restrict__ b_ih,
                       const float* __restrict__ b_hh,
                       float* __restrict__ xproj) {
    const int lane = threadIdx.x & 31;
    const int wave = threadIdx.x >> 5;              // 0..3
    const int tile = blockIdx.x * 4 + wave;         // 0 .. 262143
    const int mt = tile >> 8;                       // 0..1023  (M tiles)
    const int nt = tile & 255;                      // 0..255   (N tiles)

    // A row for this lane: M index r -> (t = r/64, b = r%64), x is [B,T,H]
    const int r  = mt * 16 + (lane & 15);
    const int ti = r >> 6;
    const int bi = r & 63;
    const unsigned short* arow = xb + ((size_t)bi * Tq + ti) * Hq;
    const int nrow = nt * 16 + (lane & 15);         // W_ih row (B column)

    v8f acc = {};
#pragma unroll 4
    for (int k0 = 0; k0 < Hq; k0 += 32) {
        v16bf a = load_frag16(arow, 0, Hq, k0, lane);
        v16bf b = load_frag16(wih, nrow, Hq, k0, lane);
        acc = __builtin_amdgcn_wmma_f32_16x16x32_bf16(
            false, a, false, b, (short)0, acc, false, false);
    }

    const int ncol = nt * 16 + (lane & 15);
    const float bias = b_ih[ncol] + b_hh[ncol];
#pragma unroll
    for (int j = 0; j < 8; ++j) {
        int mrow = mt * 16 + j + ((lane >> 4) << 3);
        xproj[(size_t)mrow * G4 + ncol] = acc[j] + bias;
    }
}

// ---------------------------------------------------------------------------
// Phase 2: one recurrent step (launched T times, graph-captured).
// Grid: 256 blocks = 4 M-tiles x 64 hidden-col groups; block = 512 thr.
// TDM DMAs the block's 16x1024 bf16 h-tile into LDS (hardware-padded rows
// for bank-conflict-free ds_load_b128 A-fragments). Wave w: gate g = w&3,
// K-quarter kq = w>>2 -> 8 WMMAs each; partials in LDS; pointwise update.
// ---------------------------------------------------------------------------
__global__ __launch_bounds__(512)
void lstm_step_kernel(const unsigned short* __restrict__ hin,   // [B,H] bf16
                      unsigned short* __restrict__ hout,        // [B,H] bf16
                      float* __restrict__ cbuf,                 // [B,H] f32
                      const float* __restrict__ xproj,          // [T,B,4H] f32
                      const unsigned short* __restrict__ whh,   // [4H,H] bf16
                      const int* __restrict__ dones,            // [B,T]
                      float* __restrict__ out,                  // out|h_f|c_f
                      int t) {
    __shared__ struct {
        unsigned short hstage[HSTAGE_SHORTS];   // 33.0 KB, TDM destination
        float tiles[4][4][16][16];              // 16 KB  [kq][gate][m][n]
    } sm;

    const int lane = threadIdx.x & 31;
    const int wave = threadIdx.x >> 5;    // 0..15
    const int g    = wave & 3;            // gate: i,f,g,o
    const int kq   = wave >> 2;           // K quarter
    const int mt   = blockIdx.x >> 6;     // 0..3   (batch tile)
    const int ng   = blockIdx.x & 63;     // 0..63  (hidden col group)

    // ---- TDM: DMA h tile [16 rows x 1024 bf16] global -> LDS (wave 0) ----
    if (wave == 0) {
        uint64_t gaddr = (uint64_t)(uintptr_t)(hin + (size_t)mt * 16 * Hq);
        uint32_t ldsa  = (uint32_t)(uintptr_t)&sm.hstage[0]; // flat low 32 = LDS offset
        u32x4 g0;
        g0[0] = 1u;                                   // count=1 (valid descriptor)
        g0[1] = ldsa;                                 // lds_addr
        g0[2] = (uint32_t)gaddr;                      // global_addr[31:0]
        g0[3] = ((uint32_t)(gaddr >> 32) & 0x01FFFFFFu) | 0x80000000u; // addr[56:32] | type=2
        i32x8 g1;
        g1[0] = (int)((1u << 16)      // data_size = 2 bytes
                    | (1u << 20)      // pad_enable
                    | (7u << 22)      // pad_interval: every 256 dwords
                    | (1u << 25));    // pad_amount: 2 dwords
        g1[1] = (int)(1024u << 16);   // tensor_dim0[15:0] = 1024
        g1[2] = (int)(16u << 16);     // tensor_dim0 hi=0 | tensor_dim1[15:0]=16
        g1[3] = (int)(1024u << 16);   // tensor_dim1 hi=0 | tile_dim0=1024
        g1[4] = 16;                   // tile_dim1=16, tile_dim2=0
        g1[5] = 1024;                 // tensor_dim0_stride lo
        g1[6] = 0;
        g1[7] = 0;
        i32x4 g2 = {0, 0, 0, 0};      // group2: unused (2D tensor)
        i32x4 g3 = {0, 0, 0, 0};      // group3: unused (2D tensor)
        i32x8 g4 = {0, 0, 0, 0, 0, 0, 0, 0};  // extra group (clang-23 6-arg form)
        __builtin_amdgcn_tensor_load_to_lds(g0, g1, g2, g3, g4, 0);
        __builtin_amdgcn_s_wait_tensorcnt(0);
    }
    // never-taken store: keeps compiler from assuming hstage is unwritten
    if (blockIdx.x == 0x7FFFFFFFu) sm.hstage[threadIdx.x] = (unsigned short)t;
    __syncthreads();

    // ---- GEMM partial: gates[16b x 16n] over K slice [kq*256, +256) ----
    const int bl = lane & 15;
    // padded LDS row base; entire kq slice sits on one side of the mid-row pad
    const char* hrow = (const char*)&sm.hstage[0] + bl * HSTRIDE_B + (kq >= 2 ? 8 : 0);
    const int nrow = g * Hq + ng * 16 + bl;                 // W_hh row
    const unsigned short* wrow = whh + (size_t)nrow * Hq;

    v8f acc = {};
#pragma unroll
    for (int kk = 0; kk < 8; ++kk) {
        int k0 = kq * 256 + kk * 32;
        int ka = k0 + ((lane >> 4) << 3);
        FragU fa;
        fa.q.lo = *reinterpret_cast<const uint4*>(hrow + 2 * ka);       // K=ka..ka+7
        fa.q.hi = *reinterpret_cast<const uint4*>(hrow + 2 * ka + 32);  // K=ka+16..+23
        v16bf b = load_frag16(wrow, 0, Hq, k0, lane);
        acc = __builtin_amdgcn_wmma_f32_16x16x32_bf16(
            false, fa.v, false, b, (short)0, acc, false, false);
    }
#pragma unroll
    for (int j = 0; j < 8; ++j) {
        int ml = j + ((lane >> 4) << 3);
        sm.tiles[kq][g][ml][lane & 15] = acc[j];
    }
    __syncthreads();

    // ---- pointwise LSTM cell: 256 elements (16 batch x 16 hidden) ----
    const int e = threadIdx.x;
    if (e < 256) {
        const int el = e >> 4, nl = e & 15;
        const int b    = mt * 16 + el;
        const int hcol = ng * 16 + nl;

        float iv = sm.tiles[0][0][el][nl] + sm.tiles[1][0][el][nl] + sm.tiles[2][0][el][nl] + sm.tiles[3][0][el][nl];
        float fv = sm.tiles[0][1][el][nl] + sm.tiles[1][1][el][nl] + sm.tiles[2][1][el][nl] + sm.tiles[3][1][el][nl];
        float gv = sm.tiles[0][2][el][nl] + sm.tiles[1][2][el][nl] + sm.tiles[2][2][el][nl] + sm.tiles[3][2][el][nl];
        float ov = sm.tiles[0][3][el][nl] + sm.tiles[1][3][el][nl] + sm.tiles[2][3][el][nl] + sm.tiles[3][3][el][nl];

        const size_t rbase = ((size_t)t * Bq + b) * G4 + hcol;
        iv += xproj[rbase + 0 * Hq];
        fv += xproj[rbase + 1 * Hq];
        gv += xproj[rbase + 2 * Hq];
        ov += xproj[rbase + 3 * Hq];

        float ig = sigm(iv);
        float fg = sigm(fv);
        float gg = tanhf(gv);
        float og = sigm(ov);

        const size_t sidx = (size_t)b * Hq + hcol;
        float c_new = fg * cbuf[sidx] + ig * gg;
        float h_new = og * tanhf(c_new);

        // emit output BEFORE reset mask (matches reference)
        out[(size_t)b * Tq * Hq + (size_t)t * Hq + hcol] = tanhf(h_new);

        float m  = dones[b * Tq + t] ? 0.0f : 1.0f;
        float hm = h_new * m;
        float cm = c_new * m;

        hout[sidx] = f2bf(hm);
        cbuf[sidx] = cm;

        if (t == Tq - 1) {
            const size_t tail = (size_t)Bq * Tq * Hq;
            out[tail + sidx]           = hm;  // h_f
            out[tail + Bq * Hq + sidx] = cm;  // c_f
        }
    }
}

// ---------------------------------------------------------------------------
// Launch
// ---------------------------------------------------------------------------
extern "C" void kernel_launch(void* const* d_in, const int* in_sizes, int n_in,
                              void* d_out, int out_size, void* d_ws, size_t ws_size,
                              hipStream_t stream) {
    const float* x    = (const float*)d_in[0];   // [B,T,H]
    const float* h0   = (const float*)d_in[1];   // [1,B,H]
    const float* c0   = (const float*)d_in[2];   // [1,B,H]
    const int*   dn   = (const int*)  d_in[3];   // [B,T]
    const float* Wih  = (const float*)d_in[4];   // [4H,H]
    const float* Whh  = (const float*)d_in[5];   // [4H,H]
    const float* bih  = (const float*)d_in[6];   // [4H]
    const float* bhh  = (const float*)d_in[7];   // [4H]
    float* out = (float*)d_out;

    // Workspace layout (bytes)
    char* ws = (char*)d_ws;
    unsigned short* x_bf   = (unsigned short*)(ws);                       // 32 MB
    unsigned short* wih_bf = (unsigned short*)(ws + 33554432);            //  8 MB
    unsigned short* whh_bf = (unsigned short*)(ws + 33554432 + 8388608);  //  8 MB
    float*          xproj  = (float*)(ws + 50331648);                     // 256 MB
    unsigned short* hbuf   = (unsigned short*)(ws + 50331648 + 268435456);// 2x128KB
    float*          cbuf   = (float*)(ws + 50331648 + 268435456 + 262144);// 256 KB

    // Phase 0: conversions + state init
    cvt_bf16_kernel<<<8192, 256, 0, stream>>>(x,   x_bf,   Bq * Tq * Hq);
    cvt_bf16_kernel<<<4096, 256, 0, stream>>>(Wih, wih_bf, G4 * Hq);
    cvt_bf16_kernel<<<4096, 256, 0, stream>>>(Whh, whh_bf, G4 * Hq);
    init_state_kernel<<<(Bq * Hq) / 256, 256, 0, stream>>>(h0, c0, hbuf, cbuf);

    // Phase 1: x_proj = x @ W_ih^T + b_ih + b_hh   (1024 x 256 tiles, 4 waves/WG)
    xproj_gemm_kernel<<<65536, 128, 0, stream>>>(x_bf, wih_bf, bih, bhh, xproj);

    // Phase 2: sequential recurrent steps, ping-pong h buffers
    const int HSZ = Bq * Hq;
    for (int t = 0; t < Tq; ++t) {
        unsigned short* hin  = hbuf + (t & 1) * HSZ;
        unsigned short* hout = hbuf + ((t + 1) & 1) * HSZ;
        lstm_step_kernel<<<256, 512, 0, stream>>>(hin, hout, cbuf, xproj,
                                                  whh_bf, dn, out, t);
    }
}
